// GNN_7447473291643
// MI455X (gfx1250) — matrix operand order repaired and checked by reading, not compiled, not run on hardware
//
#include <hip/hip_runtime.h>

#define N_NODES 100000
#define N_EDGES 1600000
#define ETOT (N_EDGES + N_NODES)
#define IN_DIM 128
#define HID 64
#define OUT_DIM 16
#define NGRAPH 100
#define NEG_SLOPE 0.2f

typedef __attribute__((ext_vector_type(16))) _Float16 v16h;
typedef __attribute__((ext_vector_type(8)))  _Float16 v8h;
typedef __attribute__((ext_vector_type(4)))  _Float16 v4h;
typedef __attribute__((ext_vector_type(8)))  float    v8f;

union Frag16 { v16h v; v8h h[2]; };

// ---------------- utility kernels ----------------

__global__ void k_zero(float* __restrict__ p, int n) {
  int i = blockIdx.x * blockDim.x + threadIdx.x;
  if (i < n) p[i] = 0.f;
}

// vectorized f32 -> f16 (4 elems / thread, b128 in, b64 out)
__global__ void k_cvt_f16v4(const float4* __restrict__ s, v4h* __restrict__ d, int n4) {
  int i = blockIdx.x * blockDim.x + threadIdx.x;
  if (i < n4) {
    float4 v = s[i];
    v4h o = { (_Float16)v.x, (_Float16)v.y, (_Float16)v.z, (_Float16)v.w };
    d[i] = o;
  }
}

// W [K, NC] f32 row-major  ->  Wt [NC, K] f16 (so B fragments load contiguously)
__global__ void k_transpose_cvt(const float* __restrict__ W, _Float16* __restrict__ Wt,
                                int K, int NC) {
  int i = blockIdx.x * blockDim.x + threadIdx.x;
  if (i < K * NC) {
    int k = i / NC, c = i % NC;
    Wt[(size_t)c * K + k] = (_Float16)W[i];
  }
}

// ---------------- WMMA GEMM: C[M,NCOLS] = A[M,128] * Bt[NCOLS,128]^T ----------------
// One wave per 16x16 tile; K=128 fully unrolled as 4 x v_wmma_f32_16x16x32_f16.
template<int NCOLS>
__global__ void __launch_bounds__(256) k_wmma_gemm(const _Float16* __restrict__ A,
                                                   const _Float16* __restrict__ Bt,
                                                   float* __restrict__ C, int M) {
  constexpr int K = 128;
  constexpr int NT = NCOLS / 16;
  int wave = (int)((blockIdx.x * blockDim.x + threadIdx.x) >> 5);
  int lane = threadIdx.x & 31;
  int mt = wave / NT, nt = wave % NT;
  if (mt * 16 >= M) return;
  int hi = lane >> 4;
  const _Float16* arow = A  + (size_t)(mt * 16 + (lane & 15)) * K;
  const _Float16* brow = Bt + (size_t)(nt * 16 + (lane & 15)) * K;
  int ak = hi * 8;
  int bk = hi * 8;
  v8f acc = {};
#pragma unroll
  for (int k0 = 0; k0 < K; k0 += 32) {
    Frag16 a, b;
    a.h[0] = *(const v8h*)(arow + k0 + ak);
    a.h[1] = *(const v8h*)(arow + k0 + 16 + ak);
    b.h[0] = *(const v8h*)(brow + k0 + bk);
    b.h[1] = *(const v8h*)(brow + k0 + 16 + bk);
    acc = __builtin_amdgcn_wmma_f32_16x16x32_f16(
        false, a.v, false, b.v, (short)0, acc, false, false);
  }
  float* cp = C + (size_t)(mt * 16 + hi * 8) * NCOLS + nt * 16 + (lane & 15);
#pragma unroll
  for (int r = 0; r < 8; ++r) cp[(size_t)r * NCOLS] = acc[r];
}

// ---------------- attention scalars: alpha_src/dst = (h * att).sum(-1) ----------------

__global__ void k_alpha(const float* __restrict__ h, const float* __restrict__ att_s,
                        const float* __restrict__ att_d, float* __restrict__ as,
                        float* __restrict__ ad, int n, int H, int C) {
  int i = blockIdx.x * blockDim.x + threadIdx.x;
  if (i >= n * H) return;
  int node = i / H, hh = i % H;
  const float4* hp = (const float4*)(h + (size_t)node * H * C + (size_t)hh * C);
  const float4* sp = (const float4*)(att_s + hh * C);
  const float4* dp = (const float4*)(att_d + hh * C);
  float s = 0.f, d = 0.f;
#pragma unroll 4
  for (int c = 0; c < C / 4; ++c) {
    float4 v = hp[c], a = sp[c], b = dp[c];
    s += v.x * a.x + v.y * a.y + v.z * a.z + v.w * a.w;
    d += v.x * b.x + v.y * b.y + v.z * b.z + v.w * b.w;
  }
  as[i] = s; ad[i] = d;
}

// ---------------- edge passes (segment softmax + scatter) ----------------

__device__ __forceinline__ unsigned fkey(float f) {        // order-preserving f32->u32
  unsigned u = __float_as_uint(f);
  return (u & 0x80000000u) ? ~u : (u | 0x80000000u);
}
__device__ __forceinline__ float finv(unsigned k) {
  unsigned u = (k & 0x80000000u) ? (k & 0x7FFFFFFFu) : ~k;
  return __uint_as_float(u);
}
__device__ __forceinline__ void edge_sd(const int* __restrict__ ei, int e, int& s, int& d) {
  if (e < N_EDGES) { s = ei[e]; d = ei[N_EDGES + e]; }
  else             { s = d = e - N_EDGES; }                // appended self-loops
}
__device__ __forceinline__ float edge_e(const float* as, const float* ad,
                                        int s, int d, int hh, int H) {
  float v = as[s * H + hh] + ad[d * H + hh];
  return v > 0.f ? v : NEG_SLOPE * v;                      // leaky_relu
}

__global__ void k_edge_max(const float* __restrict__ as, const float* __restrict__ ad,
                           const int* __restrict__ ei, unsigned* __restrict__ mkey,
                           int H, int total) {
  int i = blockIdx.x * blockDim.x + threadIdx.x;
  if (i >= total) return;
  int e = i / H, hh = i % H, s, d;
  edge_sd(ei, e, s, d);
  atomicMax(&mkey[d * H + hh], fkey(edge_e(as, ad, s, d, hh, H)));
}

__global__ void k_edge_exp(const float* __restrict__ as, const float* __restrict__ ad,
                           const int* __restrict__ ei, const unsigned* __restrict__ mkey,
                           float* __restrict__ den, float* __restrict__ exb,
                           int H, int total) {
  int i = blockIdx.x * blockDim.x + threadIdx.x;
  if (i >= total) return;
  int e = i / H, hh = i % H, s, d;
  edge_sd(ei, e, s, d);
  float m  = finv(mkey[d * H + hh]);
  float ex = __expf(edge_e(as, ad, s, d, hh, H) - m);
  exb[i] = ex;
  atomicAdd(&den[d * H + hh], ex);
}

// one thread per (edge, 4-channel group): b128 gather of h[src], 4 f32 atomics to dst.
// Channel groups never straddle heads (C=64 is a multiple of 4).
__global__ void k_edge_scatter4(const float4* __restrict__ h4, const int* __restrict__ ei,
                                const float* __restrict__ exb, const float* __restrict__ den,
                                float* __restrict__ out, int H, int C, int total4) {
  int i = blockIdx.x * blockDim.x + threadIdx.x;
  if (i >= total4) return;
  int Q = (H * C) >> 2;                    // float4 groups per edge
  int e = i / Q, g = i % Q;
  int c0 = g << 2, hh = c0 / C, s, d;
  edge_sd(ei, e, s, d);
  float alpha = exb[(size_t)e * H + hh] / den[d * H + hh];
  float4 v = h4[(size_t)s * Q + g];
  float* op = out + (size_t)d * H * C + c0;
  atomicAdd(op + 0, v.x * alpha);
  atomicAdd(op + 1, v.y * alpha);
  atomicAdd(op + 2, v.z * alpha);
  atomicAdd(op + 3, v.w * alpha);
}

// bias + relu + f16 convert, 4 elems / thread
__global__ void k_bias_relu_f16v4(const float4* __restrict__ acc, const float* __restrict__ b,
                                  v4h* __restrict__ o, int total4, int C4) {
  int i = blockIdx.x * blockDim.x + threadIdx.x;
  if (i >= total4) return;
  const float4* b4 = (const float4*)b;
  float4 v = acc[i], bb = b4[i % C4];
  float x0 = v.x + bb.x, x1 = v.y + bb.y, x2 = v.z + bb.z, x3 = v.w + bb.w;
  v4h r = { (_Float16)(x0 > 0.f ? x0 : 0.f), (_Float16)(x1 > 0.f ? x1 : 0.f),
            (_Float16)(x2 > 0.f ? x2 : 0.f), (_Float16)(x3 > 0.f ? x3 : 0.f) };
  o[i] = r;
}

// ---------------- pooling + classifier ----------------

__global__ void k_pool4(const float4* __restrict__ h4, const int* __restrict__ batch,
                        float* __restrict__ sums, float* __restrict__ cnt) {
  int i = blockIdx.x * blockDim.x + threadIdx.x;
  if (i >= N_NODES * (HID / 4)) return;
  int n = i / (HID / 4), g = i % (HID / 4);
  int b = batch[n];
  float4 v = h4[i];
  float* sp = sums + b * HID + (g << 2);
  atomicAdd(sp + 0, v.x);
  atomicAdd(sp + 1, v.y);
  atomicAdd(sp + 2, v.z);
  atomicAdd(sp + 3, v.w);
  if (g == 0) atomicAdd(&cnt[b], 1.f);
}

__global__ void k_final(const float* __restrict__ sums, const float* __restrict__ cnt,
                        const float* __restrict__ b2, const float* __restrict__ Wl,
                        const float* __restrict__ bl, float* __restrict__ out) {
  int i = blockIdx.x * blockDim.x + threadIdx.x;
  if (i >= NGRAPH * OUT_DIM) return;
  int g = i / OUT_DIM, o = i % OUT_DIM;
  float inv = 1.f / fmaxf(cnt[g], 1.f);
  float r = bl[o];
  for (int c = 0; c < HID; ++c)
    r += (sums[g * HID + c] * inv + b2[c]) * Wl[c * OUT_DIM + o];
  out[i] = r;
}

// ---------------- launch ----------------

static inline unsigned int blks(long long n, int t) { return (unsigned int)((n + t - 1) / t); }

extern "C" void kernel_launch(void* const* d_in, const int* in_sizes, int n_in,
                              void* d_out, int out_size, void* d_ws, size_t ws_size,
                              hipStream_t stream) {
  const float* x    = (const float*)d_in[0];
  const int*   ei   = (const int*)  d_in[1];
  const int*   batch= (const int*)  d_in[2];
  const float* W1   = (const float*)d_in[3];
  const float* as1v = (const float*)d_in[4];
  const float* ad1v = (const float*)d_in[5];
  const float* b1   = (const float*)d_in[6];
  const float* W2   = (const float*)d_in[7];
  const float* as2v = (const float*)d_in[8];
  const float* ad2v = (const float*)d_in[9];
  const float* b2   = (const float*)d_in[10];
  const float* Wl   = (const float*)d_in[11];
  const float* bl   = (const float*)d_in[12];
  float* out = (float*)d_out;

  char* p = (char*)d_ws;
  auto bump = [&](size_t bytes) -> char* {
    char* r = p; p += (bytes + 255) & ~(size_t)255; return r;
  };
  _Float16* xh   = (_Float16*)bump((size_t)N_NODES * IN_DIM * 2);
  _Float16* Wt1  = (_Float16*)bump((size_t)128 * 128 * 2);
  _Float16* Wt2  = (_Float16*)bump((size_t)64 * 128 * 2);
  float*    h1   = (float*)   bump((size_t)N_NODES * 128 * 4);
  float*    as1  = (float*)   bump((size_t)N_NODES * 2 * 4);
  float*    ad1  = (float*)   bump((size_t)N_NODES * 2 * 4);
  unsigned* mk1  = (unsigned*)bump((size_t)N_NODES * 2 * 4);
  float*    den1 = (float*)   bump((size_t)N_NODES * 2 * 4);
  float*    ex1  = (float*)   bump((size_t)ETOT * 2 * 4);
  float*    out1 = (float*)   bump((size_t)N_NODES * 128 * 4);
  _Float16* h1h  = (_Float16*)bump((size_t)N_NODES * 128 * 2);
  float*    h2   = (float*)   bump((size_t)N_NODES * 64 * 4);
  float*    as2  = (float*)   bump((size_t)N_NODES * 4);
  float*    ad2  = (float*)   bump((size_t)N_NODES * 4);
  unsigned* mk2  = (unsigned*)bump((size_t)N_NODES * 4);
  float*    den2 = (float*)   bump((size_t)N_NODES * 4);
  float*    ex2  = (float*)   bump((size_t)ETOT * 4);
  float*    out2 = (float*)   bump((size_t)N_NODES * 64 * 4);
  float*    sums = (float*)   bump((size_t)NGRAPH * HID * 4);
  float*    cnt  = (float*)   bump((size_t)NGRAPH * 4);

  const int T = 256;

  // precision prep: fp16 copies of activations, transposed fp16 weights
  k_cvt_f16v4<<<blks((long long)N_NODES * IN_DIM / 4, T), T, 0, stream>>>(
      (const float4*)x, (v4h*)xh, N_NODES * IN_DIM / 4);
  k_transpose_cvt<<<blks(128 * 128, T), T, 0, stream>>>(W1, Wt1, 128, 128);
  k_transpose_cvt<<<blks(128 * 64, T), T, 0, stream>>>(W2, Wt2, 128, 64);

  // zero all accumulators every call (workspace is never re-initialized by harness)
  k_zero<<<blks((long long)N_NODES * 128, T), T, 0, stream>>>(out1, N_NODES * 128);
  k_zero<<<blks((long long)N_NODES * 2, T), T, 0, stream>>>((float*)mk1, N_NODES * 2);
  k_zero<<<blks((long long)N_NODES * 2, T), T, 0, stream>>>(den1, N_NODES * 2);
  k_zero<<<blks((long long)N_NODES * 64, T), T, 0, stream>>>(out2, N_NODES * 64);
  k_zero<<<blks(N_NODES, T), T, 0, stream>>>((float*)mk2, N_NODES);
  k_zero<<<blks(N_NODES, T), T, 0, stream>>>(den2, N_NODES);
  k_zero<<<blks(NGRAPH * HID, T), T, 0, stream>>>(sums, NGRAPH * HID);
  k_zero<<<blks(NGRAPH, T), T, 0, stream>>>(cnt, NGRAPH);

  // ---- layer 1: h1 = x @ W1 (WMMA), attention, scatter, bias+relu ----
  {
    long long waves = (long long)(N_NODES / 16) * (128 / 16);
    k_wmma_gemm<128><<<blks(waves * 32, T), T, 0, stream>>>(xh, Wt1, h1, N_NODES);
  }
  k_alpha<<<blks((long long)N_NODES * 2, T), T, 0, stream>>>(h1, as1v, ad1v, as1, ad1,
                                                            N_NODES, 2, 64);
  k_edge_max<<<blks((long long)ETOT * 2, T), T, 0, stream>>>(as1, ad1, ei, mk1, 2, ETOT * 2);
  k_edge_exp<<<blks((long long)ETOT * 2, T), T, 0, stream>>>(as1, ad1, ei, mk1, den1, ex1,
                                                            2, ETOT * 2);
  k_edge_scatter4<<<blks((long long)ETOT * 32, T), T, 0, stream>>>(
      (const float4*)h1, ei, ex1, den1, out1, 2, 64, ETOT * 32);
  k_bias_relu_f16v4<<<blks((long long)N_NODES * 32, T), T, 0, stream>>>(
      (const float4*)out1, b1, (v4h*)h1h, N_NODES * 32, 32);

  // ---- layer 2: h2 = relu(h1) @ W2 (WMMA), attention, scatter ----
  {
    long long waves = (long long)(N_NODES / 16) * (64 / 16);
    k_wmma_gemm<64><<<blks(waves * 32, T), T, 0, stream>>>(h1h, Wt2, h2, N_NODES);
  }
  k_alpha<<<blks(N_NODES, T), T, 0, stream>>>(h2, as2v, ad2v, as2, ad2, N_NODES, 1, 64);
  k_edge_max<<<blks((long long)ETOT, T), T, 0, stream>>>(as2, ad2, ei, mk2, 1, ETOT);
  k_edge_exp<<<blks((long long)ETOT, T), T, 0, stream>>>(as2, ad2, ei, mk2, den2, ex2, 1, ETOT);
  k_edge_scatter4<<<blks((long long)ETOT * 16, T), T, 0, stream>>>(
      (const float4*)h2, ei, ex2, den2, out2, 1, 64, ETOT * 16);

  // ---- mean pool + classifier (b2 folded in post-mean; it is per-node constant) ----
  k_pool4<<<blks((long long)N_NODES * (HID / 4), T), T, 0, stream>>>(
      (const float4*)out2, batch, sums, cnt);
  k_final<<<blks(NGRAPH * OUT_DIM, T), T, 0, stream>>>(sums, cnt, b2, Wl, bl, out);
}
